// GPTOSSAttentionSDPABlock_27547920237058
// MI455X (gfx1250) — matrix operand order repaired and checked by reading, not compiled, not run on hardware
//
#include <hip/hip_runtime.h>
#include <math.h>

#define N_TOKENS 2048
#define KV_HEADS 8
#define Q_MULT   8
#define HEAD_DIM 64
#define HIDDEN   2880
#define WINDOW   128
#define NHEADS   64     /* KV_HEADS * Q_MULT */
#define DMODEL   4096   /* NHEADS * HEAD_DIM */
#define KVDIM    512    /* KV_HEADS * HEAD_DIM */
#define SM_SCALE 0.125f /* 1/sqrt(64) */

typedef __attribute__((ext_vector_type(16))) _Float16 v16h;
typedef __attribute__((ext_vector_type(8)))  _Float16 v8h;
typedef __attribute__((ext_vector_type(8)))  float    v8f;

__device__ __forceinline__ v8f wmma_f16(v16h a, v16h b, v8f c) {
  // (neg_a, A, neg_b, B, c_mod, C, reuse_a, reuse_b)
  return __builtin_amdgcn_wmma_f32_16x16x32_f16(false, a, false, b, (short)0, c,
                                                false, false);
}

__device__ __forceinline__ v16h concat8(v8h lo, v8h hi) {
  v16h r;
#pragma unroll
  for (int j = 0; j < 8; ++j) { r[j] = lo[j]; r[j + 8] = hi[j]; }
  return r;
}

// ---------------- pre-pass: f32 -> f16 conversions ----------------
__global__ __launch_bounds__(256) void cvtw_kernel(const float* __restrict__ W,
                                                   _Float16* __restrict__ Wh) {
  const size_t i = ((size_t)blockIdx.x * blockDim.x + threadIdx.x) * 8;
  const float4 x = ((const float4*)(W + i))[0];
  const float4 y = ((const float4*)(W + i))[1];
  v8h o;
  o[0] = (_Float16)x.x; o[1] = (_Float16)x.y; o[2] = (_Float16)x.z; o[3] = (_Float16)x.w;
  o[4] = (_Float16)y.x; o[5] = (_Float16)y.y; o[6] = (_Float16)y.z; o[7] = (_Float16)y.w;
  *(v8h*)(Wh + i) = o;
}

__global__ __launch_bounds__(256) void cvtk_kernel(const float* __restrict__ k,
                                                   _Float16* __restrict__ kh) {
  const size_t i = ((size_t)blockIdx.x * blockDim.x + threadIdx.x) * 8;
  const float4 x = ((const float4*)(k + i))[0];
  const float4 y = ((const float4*)(k + i))[1];
  v8h o;
  o[0] = (_Float16)x.x; o[1] = (_Float16)x.y; o[2] = (_Float16)x.z; o[3] = (_Float16)x.w;
  o[4] = (_Float16)y.x; o[5] = (_Float16)y.y; o[6] = (_Float16)y.z; o[7] = (_Float16)y.w;
  *(v8h*)(kh + i) = o;
}

// v [2048,8,64] f32 -> vt [8,64,2048] f16 (transpose so V^T A-operand is contiguous)
__global__ __launch_bounds__(256) void cvtvt_kernel(const float* __restrict__ v,
                                                    _Float16* __restrict__ vt) {
  const int tid  = blockIdx.x * 256 + threadIdx.x;  // 131072 threads
  const int hd   = tid >> 8;                        // 0..511 = h*64+d
  const int tok0 = (tid & 255) * 8;
  v8h o;
#pragma unroll
  for (int j = 0; j < 8; ++j)
    o[j] = (_Float16)v[(size_t)(tok0 + j) * KVDIM + hd];
  *(v8h*)(vt + (size_t)hd * N_TOKENS + tok0) = o;
}

// ------- Attention (transposed): one wave per (16-query tile, head) -------
// Block mapping: q0 = (blockIdx>>3)*16 (block-uniform), h = blockIdx&7
// (block-uniform), mult = wave id -> all 8 waves share the same K/V slices.
// S^T = K (Q*scale)^T ; O^T = V^T P^T.  Lane = query column; per-lane scalar
// softmax state; no LDS round-trip (packed half-wave shfl builds P^T).
// mask mode: 0 = none, 1 = causal only (key <= qq), 2 = window only (key > qq-128)
__global__ __launch_bounds__(256) void attn_kernel(const float* __restrict__ q,
                                                   const _Float16* __restrict__ kh,
                                                   const _Float16* __restrict__ vt,
                                                   const float* __restrict__ sinks,
                                                   _Float16* __restrict__ t) {
  const int lane = threadIdx.x & 31;
  const int wid  = threadIdx.x >> 5;           // = q_mult index
  const int h    = blockIdx.x & 7;             // block-uniform kv head
  const int q0   = (blockIdx.x >> 3) * 16;     // block-uniform query tile
  const int head = h * Q_MULT + wid;
  const int col  = lane & 15;
  const int hig  = (lane >> 4) & 1;
  const int ab0  = (lane & 16) ? 8 : 0;        // unused; A elements also use bb0 (one 16-seg)
  const int bb0  = (lane & 16) ? 16 : 0;       // B element K-dim base (one 16-seg)
  const int qq   = q0 + col;                   // this lane's query row
  const int qlo  = qq - WINDOW;                // window: key > qlo

  // Q^T as B operand (scale folded): element e -> d = bb0 + e (+32 for chunk 1)
  const float* qrow = q + (size_t)qq * DMODEL + head * HEAD_DIM;
  v16h bq0, bq1;
#pragma unroll
  for (int j = 0; j < 16; ++j) {
    bq0[j] = (_Float16)(qrow[bb0 + j] * SM_SCALE);
    bq1[j] = (_Float16)(qrow[32 + bb0 + j] * SM_SCALE);
  }

  float mrun = -1e30f, lrun = 0.f;
  v8f ot0 = {}, ot1 = {}, ot2 = {}, ot3 = {};

  int kstart = q0 - (WINDOW - 1);
  if (kstart < 0) kstart = 0;
  kstart &= ~31;

  // V^T A-operand rows: d = dtile*16 + col, keys contiguous (shared across waves)
  const _Float16* vr0 = vt + (size_t)(h * HEAD_DIM + 0  + col) * N_TOKENS;
  const _Float16* vr1 = vt + (size_t)(h * HEAD_DIM + 16 + col) * N_TOKENS;
  const _Float16* vr2 = vt + (size_t)(h * HEAD_DIM + 32 + col) * N_TOKENS;
  const _Float16* vr3 = vt + (size_t)(h * HEAD_DIM + 48 + col) * N_TOKENS;

  auto chunk = [&](int kc, int mode) {
    // ---- S^T: A = K rows (lane = key), B = Q^T ----
    const _Float16* kr0 = kh + (size_t)(kc + col) * KVDIM + h * HEAD_DIM;
    const _Float16* kr1 = kh + (size_t)(kc + 16 + col) * KVDIM + h * HEAD_DIM;
    const v16h ak0lo = concat8(*(const v8h*)(kr0 + bb0), *(const v8h*)(kr0 + bb0 + 8));
    const v16h ak0hi = concat8(*(const v8h*)(kr0 + 32 + bb0), *(const v8h*)(kr0 + 32 + bb0 + 8));
    const v16h ak1lo = concat8(*(const v8h*)(kr1 + bb0), *(const v8h*)(kr1 + bb0 + 8));
    const v16h ak1hi = concat8(*(const v8h*)(kr1 + 32 + bb0), *(const v8h*)(kr1 + 32 + bb0 + 8));
    v8f st0 = {}, st1 = {};
    st0 = wmma_f16(ak0lo, bq0, st0);
    st0 = wmma_f16(ak0hi, bq1, st0);
    st1 = wmma_f16(ak1lo, bq0, st1);
    st1 = wmma_f16(ak1hi, bq1, st1);
    // st0[r]: key = kc + 8*hig + r ; st1[r]: key = kc+16 + 8*hig + r

    float x0[8], x1[8];
    bool v0[8], v1[8];
#pragma unroll
    for (int r = 0; r < 8; ++r) {
      x0[r] = st0[r]; x1[r] = st1[r];
      v0[r] = true; v1[r] = true;
      if (mode) {
        const int k0i = kc + 8 * hig + r;
        const int k1i = kc + 16 + 8 * hig + r;
        v0[r] = (mode == 1) ? (k0i <= qq) : (k0i > qlo);
        v1[r] = (mode == 1) ? (k1i <= qq) : (k1i > qlo);
        x0[r] = v0[r] ? x0[r] : -1e30f;
        x1[r] = v1[r] ? x1[r] : -1e30f;
      }
    }
    // chunk max: in-lane over 16, then partner half-wave
    float m = fmaxf(x0[0], x1[0]);
#pragma unroll
    for (int r = 1; r < 8; ++r) m = fmaxf(m, fmaxf(x0[r], x1[r]));
    m = fmaxf(m, __shfl_xor(m, 16, 32));
    const float mnew  = fmaxf(mrun, m);
    const float alpha = __expf(mrun - mnew);
    mrun = mnew;

    float p0[8], p1[8], psum = 0.f;
#pragma unroll
    for (int r = 0; r < 8; ++r) {
      p0[r] = __expf(x0[r] - mnew);
      p1[r] = __expf(x1[r] - mnew);
      if (mode) { p0[r] = v0[r] ? p0[r] : 0.f; p1[r] = v1[r] ? p1[r] : 0.f; }
      psum += p0[r] + p1[r];
    }
    psum += __shfl_xor(psum, 16, 32);
    lrun = lrun * alpha + psum;
#pragma unroll
    for (int r = 0; r < 8; ++r) {
      ot0[r] *= alpha; ot1[r] *= alpha; ot2[r] *= alpha; ot3[r] *= alpha;
    }

    // ---- build P^T B operand: convert to f16, exchange packed dwords ----
    union { v8h h; unsigned u[4]; } h0, h1;
#pragma unroll
    for (int r = 0; r < 8; ++r) { h0.h[r] = (_Float16)p0[r]; h1.h[r] = (_Float16)p1[r]; }
    unsigned s0w[4], s1w[4];
#pragma unroll
    for (int j = 0; j < 4; ++j) {
      s0w[j] = (unsigned)__shfl_xor((int)h0.u[j], 16, 32);
      s1w[j] = (unsigned)__shfl_xor((int)h1.u[j], 16, 32);
    }
    // element e -> key = kc + 16*hig + e
    union { v16h h; unsigned u[8]; } pbu;
#pragma unroll
    for (int j = 0; j < 4; ++j) {
      pbu.u[j]     = hig ? s1w[j] : h0.u[j];  // keys kc+16hig + 0..7
      pbu.u[4 + j] = hig ? h1.u[j] : s0w[j];  // keys kc+16hig + 8..15
    }
    const v16h pb = pbu.h;

    // ---- O^T += V^T P^T (four d-tiles of 16 rows) ----
    const v16h av0 = concat8(*(const v8h*)(vr0 + kc + bb0), *(const v8h*)(vr0 + kc + bb0 + 8));
    const v16h av1 = concat8(*(const v8h*)(vr1 + kc + bb0), *(const v8h*)(vr1 + kc + bb0 + 8));
    const v16h av2 = concat8(*(const v8h*)(vr2 + kc + bb0), *(const v8h*)(vr2 + kc + bb0 + 8));
    const v16h av3 = concat8(*(const v8h*)(vr3 + kc + bb0), *(const v8h*)(vr3 + kc + bb0 + 8));
    ot0 = wmma_f16(av0, pb, ot0);
    ot1 = wmma_f16(av1, pb, ot1);
    ot2 = wmma_f16(av2, pb, ot2);
    ot3 = wmma_f16(av3, pb, ot3);
  };

  // first chunk: window edge (q0>=128), causal edge (q0<32), else unmasked.
  // mode is block-uniform -> scalar branch, no EXEC masking.
  int kc = kstart;
  const int firstmode = (q0 >= WINDOW) ? 2 : ((q0 < 32) ? 1 : 0);
  chunk(kc, firstmode);
  kc += 32;
  // interior chunks: fully valid (second chunk starts at q0-96 or q0-112 >= qq_max-127)
  for (; kc <= q0 - 32; kc += 32) chunk(kc, 0);
  // tail chunk(s): causal edge only (keys >= q0-16 > qq-128 always)
  for (; kc <= q0 + 15; kc += 32) chunk(kc, 1);

  // ---- sink logit, normalize, vectorized f16 store ----
  const float sv = sinks[head];
  const float M = fmaxf(mrun, sv);
  const float a = __expf(mrun - M);
  const float l = lrun * a + __expf(sv - M);
  const float scl = a / l;

  v8h s0h, s1h, s2h, s3h;
#pragma unroll
  for (int r = 0; r < 8; ++r) {
    s0h[r] = (_Float16)(ot0[r] * scl);
    s1h[r] = (_Float16)(ot1[r] * scl);
    s2h[r] = (_Float16)(ot2[r] * scl);
    s3h[r] = (_Float16)(ot3[r] * scl);
  }
  _Float16* trow = t + (size_t)qq * DMODEL + head * HEAD_DIM + 8 * hig;
  *(v8h*)(trow + 0)  = s0h;   // d = 16*tile + 8*hig + r
  *(v8h*)(trow + 16) = s1h;
  *(v8h*)(trow + 32) = s2h;
  *(v8h*)(trow + 48) = s3h;
}

// -------- Projection: out = t @ Wh^T + b ; one wave per 32x64 tile --------
__global__ __launch_bounds__(256) void proj_kernel(const _Float16* __restrict__ t,
                                                   const _Float16* __restrict__ Wh,
                                                   const float* __restrict__ bo,
                                                   float* __restrict__ out) {
  const int lane  = threadIdx.x & 31;
  const int wid   = threadIdx.x >> 5;
  const int gw    = blockIdx.x * 8 + wid;      // 2880 waves total
  const int ntile = gw % (HIDDEN / 64);        // 45
  const int mtile = gw / (HIDDEN / 64);        // 64
  const int m0 = mtile * 32, n0 = ntile * 64;
  const int col = lane & 15;
  const int hig = (lane >> 4) & 1;

  v8f a0c0 = {}, a0c1 = {}, a0c2 = {}, a0c3 = {};
  v8f a1c0 = {}, a1c1 = {}, a1c2 = {}, a1c3 = {};
  const _Float16* trow0 = t + (size_t)(m0 + col) * DMODEL;
  const _Float16* trow1 = t + (size_t)(m0 + 16 + col) * DMODEL;
  const _Float16* w0 = Wh + (size_t)(n0 + col) * DMODEL;
  const _Float16* w1 = Wh + (size_t)(n0 + 16 + col) * DMODEL;
  const _Float16* w2 = Wh + (size_t)(n0 + 32 + col) * DMODEL;
  const _Float16* w3 = Wh + (size_t)(n0 + 48 + col) * DMODEL;
  const int ab0 = (lane & 16) ? 8 : 0;   // unused; A elements also use bb0 (one 16-seg)
  const int bb0 = (lane & 16) ? 16 : 0;  // B element K base (one 16-seg)

  for (int k0 = 0; k0 < DMODEL; k0 += 32) {
    const v16h a0 = concat8(*(const v8h*)(trow0 + k0 + bb0),
                            *(const v8h*)(trow0 + k0 + bb0 + 8));
    const v16h a1 = concat8(*(const v8h*)(trow1 + k0 + bb0),
                            *(const v8h*)(trow1 + k0 + bb0 + 8));
    const v16h b0v = *(const v16h*)(w0 + k0 + bb0);  // 32B aligned contiguous
    const v16h b1v = *(const v16h*)(w1 + k0 + bb0);
    const v16h b2v = *(const v16h*)(w2 + k0 + bb0);
    const v16h b3v = *(const v16h*)(w3 + k0 + bb0);
    a0c0 = wmma_f16(a0, b0v, a0c0);
    a1c0 = wmma_f16(a1, b0v, a1c0);
    a0c1 = wmma_f16(a0, b1v, a0c1);
    a1c1 = wmma_f16(a1, b1v, a1c1);
    a0c2 = wmma_f16(a0, b2v, a0c2);
    a1c2 = wmma_f16(a1, b2v, a1c2);
    a0c3 = wmma_f16(a0, b3v, a0c3);
    a1c3 = wmma_f16(a1, b3v, a1c3);
    __builtin_prefetch(w0 + k0 + 1024, 0, 1);
    __builtin_prefetch(w1 + k0 + 1024, 0, 1);
    __builtin_prefetch(w2 + k0 + 1024, 0, 1);
    __builtin_prefetch(w3 + k0 + 1024, 0, 1);
  }

  const float bv0 = bo[n0 + col];
  const float bv1 = bo[n0 + 16 + col];
  const float bv2 = bo[n0 + 32 + col];
  const float bv3 = bo[n0 + 48 + col];
#pragma unroll
  for (int r = 0; r < 8; ++r) {
    float* orow0 = out + (size_t)(m0 + 8 * hig + r) * HIDDEN + n0;
    orow0[col]      = a0c0[r] + bv0;
    orow0[16 + col] = a0c1[r] + bv1;
    orow0[32 + col] = a0c2[r] + bv2;
    orow0[48 + col] = a0c3[r] + bv3;
    float* orow1 = out + (size_t)(m0 + 16 + 8 * hig + r) * HIDDEN + n0;
    orow1[col]      = a1c0[r] + bv0;
    orow1[16 + col] = a1c1[r] + bv1;
    orow1[32 + col] = a1c2[r] + bv2;
    orow1[48 + col] = a1c3[r] + bv3;
  }
}

extern "C" void kernel_launch(void* const* d_in, const int* in_sizes, int n_in,
                              void* d_out, int out_size, void* d_ws, size_t ws_size,
                              hipStream_t stream) {
  (void)in_sizes; (void)n_in; (void)out_size; (void)ws_size;
  const float* q     = (const float*)d_in[0];
  const float* k     = (const float*)d_in[1];
  const float* v     = (const float*)d_in[2];
  const float* sinks = (const float*)d_in[3];
  const float* W     = (const float*)d_in[4];
  const float* bo    = (const float*)d_in[5];
  float* out = (float*)d_out;

  char* ws = (char*)d_ws;
  _Float16* t  = (_Float16*)ws;                                   // 16.78 MB
  _Float16* Wh = (_Float16*)(ws + (size_t)N_TOKENS * DMODEL * 2); // 23.59 MB
  _Float16* kh = (_Float16*)(ws + (size_t)N_TOKENS * DMODEL * 2
                                + (size_t)HIDDEN * DMODEL * 2);   // 2.10 MB
  _Float16* vt = kh + (size_t)N_TOKENS * KVDIM;                   // 2.10 MB

  // pre-conversions (f32 -> f16)
  cvtw_kernel<<<5760, 256, 0, stream>>>(W, Wh);   // 2880*4096 / 8 / 256
  cvtk_kernel<<<512, 256, 0, stream>>>(k, kh);    // 2048*512 / 8 / 256
  cvtvt_kernel<<<512, 256, 0, stream>>>(v, vt);   // 512 cols * 256 tok-chunks
  // blockIdx = qtile*8 + kv_head ; wave = q_mult -> 1024 blocks
  attn_kernel<<<1024, 256, 0, stream>>>(q, kh, vt, sinks, t);
  // 64 mtiles * 45 ntiles = 2880 waves, 8 waves/block
  proj_kernel<<<360, 256, 0, stream>>>(t, Wh, bo, out);
}
